// VQEncoder_15504831939336
// MI455X (gfx1250) — matrix operand order repaired and checked by reading, not compiled, not run on hardware
//
#include <hip/hip_runtime.h>
#include <math.h>

// ---------------------------------------------------------------------------
// Types for CDNA5 WMMA (wave32): A/B = 16 x bf16, C/D = 8 x f32
// ---------------------------------------------------------------------------
typedef float  f32x4 __attribute__((ext_vector_type(4)));
typedef float  v8f   __attribute__((ext_vector_type(8)));
typedef __bf16 v16bf __attribute__((ext_vector_type(16)));

union Frag {
    v16bf          v;
    f32x4          f4[2];
    unsigned short u[16];
};

__device__ __forceinline__ unsigned short f2bf(float f) {
    unsigned int u = __float_as_uint(f);
    unsigned int r = u + 0x7FFFu + ((u >> 16) & 1u);   // RNE
    return (unsigned short)(r >> 16);
}

__device__ __forceinline__ float sigf(float x) { return 1.0f / (1.0f + __expf(-x)); }

__device__ __forceinline__ v8f wmma_bf16(const Frag& a, const Frag& b, v8f c) {
    return __builtin_amdgcn_wmma_f32_16x16x32_bf16(false, a.v, false, b.v,
                                                   (short)0, c, false, false);
}

#define NPOS 32768   // B*T = 32*1024

// ---------------------------------------------------------------------------
// Packing / folding kernels
// ---------------------------------------------------------------------------
__global__ void pack_x0_kernel(const float* __restrict__ x,
                               const float* __restrict__ c_src,
                               unsigned short* __restrict__ X0) {
    size_t gid = (size_t)blockIdx.x * 256 + threadIdx.x;
    if (gid >= (size_t)352 * NPOS) return;
    int ci = (int)(gid / NPOS);
    int n  = (int)(gid % NPOS);
    int b = n >> 10, t = n & 1023;
    float v = 0.0f;
    if (ci < 80)       v = x[((size_t)b * 80 + ci) * 1024 + t];
    else if (ci < 336) v = c_src[((size_t)b * 256 + (ci - 80)) * 1024 + t];
    X0[gid] = f2bf(v);
}

// W (M, CinReal, 5) -> Wp[kw][M][CinPad] bf16, zero-padded in ci
__global__ void pack_convw_kernel(const float* __restrict__ W,
                                  unsigned short* __restrict__ Wp,
                                  int M, int CinReal, int CinPad) {
    size_t gid   = (size_t)blockIdx.x * 256 + threadIdx.x;
    size_t plane = (size_t)M * CinPad;
    if (gid >= (size_t)5 * plane) return;
    int kw = (int)(gid / plane);
    size_t rem = gid % plane;
    int co = (int)(rem / CinPad);
    int ci = (int)(rem % CinPad);
    float v = (ci < CinReal) ? W[((size_t)co * CinReal + ci) * 5 + kw] : 0.0f;
    Wp[gid] = f2bf(v);
}

__global__ void bnfold_kernel(const float* __restrict__ cb, const float* __restrict__ g,
                              const float* __restrict__ be, const float* __restrict__ m,
                              const float* __restrict__ v,
                              float* __restrict__ scale, float* __restrict__ bias) {
    int i = blockIdx.x * 256 + threadIdx.x;
    if (i >= 512) return;
    float s = g[i] * rsqrtf(v[i] + 1e-5f);
    scale[i] = s;
    bias[i]  = (cb[i] - m[i]) * s + be[i];
}

// wih (2,128,Cin) -> bf16 [256][Cin]; bias (2,128) -> [256]
__global__ void pack_wih_kernel(const float* __restrict__ wih, const float* __restrict__ lb,
                                unsigned short* __restrict__ Wp, float* __restrict__ bias,
                                int Cin) {
    int gid = blockIdx.x * 256 + threadIdx.x;
    int total = 256 * Cin;
    if (gid < total) Wp[gid] = f2bf(wih[gid]);
    if (gid < 256)   bias[gid] = lb[gid];
}

// codebooks -> bf16 [2][512][32]; cnorm[d][c] = 0.5*||c||^2
__global__ void pack_cb_kernel(const float* __restrict__ cb_f, const float* __restrict__ cb_b,
                               unsigned short* __restrict__ CBb, float* __restrict__ cnorm) {
    int gid = blockIdx.x * 256 + threadIdx.x;
    if (gid < 2 * 512 * 32) {
        int d = gid >> 14;
        int rem = gid & 16383;
        const float* cb = d ? cb_b : cb_f;
        CBb[gid] = f2bf(cb[rem]);
    }
    if (gid < 1024) {
        int d = gid >> 9, c = gid & 511;
        const float* cb = d ? cb_b : cb_f;
        float s = 0.0f;
        #pragma unroll
        for (int j = 0; j < 32; ++j) { float e = cb[c * 32 + j]; s += e * e; }
        cnorm[gid] = 0.5f * s;
    }
}

// ---------------------------------------------------------------------------
// Conv1d implicit-GEMM (K=5, pad 2) with haloed B tile.
// out[co][n] = relu(scale[co]*conv + bias[co]) as bf16.
// Block = 256 thr (8 waves), tile 128(M) x 128(N), K-chunk 32, all 5 taps
// consumed from one haloed LDS tile (132 cols) per chunk.
// ---------------------------------------------------------------------------
__global__ __launch_bounds__(256) void conv_gemm_kernel(
        const unsigned short* __restrict__ Wp,   // [5][M][Cin] bf16
        const unsigned short* __restrict__ Xin,  // [Cin][N] bf16
        unsigned short* __restrict__ outp,       // [M][N] bf16
        const float* __restrict__ scale, const float* __restrict__ bias,
        int M, int Cin) {
    const int N = NPOS;
    __shared__ __align__(16) unsigned short As[5 * 128 * 32];  // [kw][m][k]
    __shared__ __align__(16) unsigned short Bs[132 * 32];      // [n+2][k] transposed, halo 2

    int tid  = threadIdx.x;
    int lane = tid & 31;
    int wave = tid >> 5;
    int waveM = wave >> 1;   // 0..3 -> 32 rows
    int waveN = wave & 1;    // 0..1 -> 64 cols
    int m0 = blockIdx.y * 128;
    int n0 = blockIdx.x * 128;
    int t0 = n0 & 1023;      // 128 | 1024 -> tile inside one batch row

    v8f acc[2][4];
    #pragma unroll
    for (int mt = 0; mt < 2; ++mt)
        #pragma unroll
        for (int nt = 0; nt < 4; ++nt) acc[mt][nt] = (v8f){};

    const int ar = tid >> 1;            // A row 0..127
    const int ak = (tid & 1) * 16;      // 16-wide half of the 32-k chunk
    const int bk = tid & 31;            // B k index
    const int bseg = (tid >> 5) * 16;   // B interior col segment base

    for (int cb = 0; cb < Cin; cb += 32) {
        // ---- load A for all 5 taps (registers first) ----
        f32x4 wv[5][2];
        #pragma unroll
        for (int kw = 0; kw < 5; ++kw) {
            const f32x4* ws = (const f32x4*)(Wp + (((size_t)kw * M + m0 + ar) * Cin + cb + ak));
            wv[kw][0] = ws[0];
            wv[kw][1] = ws[1];
        }
        // ---- load B interior: 16 contiguous cols at row (cb+bk) ----
        const unsigned short* xrow = Xin + (size_t)(cb + bk) * N + n0;
        f32x4 bv0 = ((const f32x4*)(xrow + bseg))[0];
        f32x4 bv1 = ((const f32x4*)(xrow + bseg))[1];
        if (cb + 32 < Cin)  // prefetch next chunk's row (global_prefetch_b8)
            __builtin_prefetch(Xin + (size_t)(cb + 32 + bk) * N + n0 + bseg, 0, 1);
        // ---- halo columns (local 0,1,130,131), first 128 threads ----
        unsigned short hval = 0;
        int hn = 0;
        if (tid < 128) {
            int hcol = tid >> 5;                    // 0..3
            hn = (hcol < 2) ? hcol : 128 + hcol;    // 0,1,130,131
            int tt = t0 + hn - 2;
            if ((unsigned)tt < 1024u)
                hval = Xin[(size_t)(cb + (tid & 31)) * N + n0 + hn - 2];
        }

        __syncthreads();   // previous chunk's fragment reads done
        #pragma unroll
        for (int kw = 0; kw < 5; ++kw) {
            f32x4* dst = (f32x4*)(As + ((size_t)kw * 128 + ar) * 32 + ak);
            dst[0] = wv[kw][0];
            dst[1] = wv[kw][1];
        }
        {   // scatter interior cols into transposed Bs (+2 halo offset)
            union { f32x4 f[2]; unsigned short u[16]; } bu;
            bu.f[0] = bv0; bu.f[1] = bv1;
            #pragma unroll
            for (int i = 0; i < 16; ++i) Bs[(2 + bseg + i) * 32 + bk] = bu.u[i];
        }
        if (tid < 128) Bs[hn * 32 + (tid & 31)] = hval;
        __syncthreads();

        // ---- 5 taps x (2x4) WMMA tiles ----
        const int lo = (lane < 16) ? 0 : 16;
        const int ko = (lane < 16) ? 0 : 16;
        #pragma unroll
        for (int kw = 0; kw < 5; ++kw) {
            Frag afr[2];
            #pragma unroll
            for (int mt = 0; mt < 2; ++mt) {
                int m = waveM * 32 + mt * 16 + (lane & 15);
                const unsigned short* arow = As + ((size_t)kw * 128 + m) * 32;
                afr[mt].f4[0] = *(const f32x4*)(arow + lo);
                afr[mt].f4[1] = *(const f32x4*)(arow + lo + 8);
            }
            #pragma unroll
            for (int nt = 0; nt < 4; ++nt) {
                Frag bfr;
                int nr = waveN * 64 + nt * 16 + (lane & 15) + kw;  // haloed row
                bfr.f4[0] = *(const f32x4*)(Bs + nr * 32 + ko);
                bfr.f4[1] = *(const f32x4*)(Bs + nr * 32 + ko + 8);
                #pragma unroll
                for (int mt = 0; mt < 2; ++mt)
                    acc[mt][nt] = wmma_bf16(afr[mt], bfr, acc[mt][nt]);
            }
        }
    }

    // ---- epilogue: fused BN + ReLU -> bf16 ----
    #pragma unroll
    for (int mt = 0; mt < 2; ++mt)
        #pragma unroll
        for (int nt = 0; nt < 4; ++nt) {
            int gn = n0 + waveN * 64 + nt * 16 + (lane & 15);
            #pragma unroll
            for (int r = 0; r < 8; ++r) {
                int gm = m0 + waveM * 32 + mt * 16 + ((lane < 16) ? r : 8 + r);
                float v = fmaxf(acc[mt][nt][r] * scale[gm] + bias[gm], 0.0f);
                outp[(size_t)gm * N + gn] = f2bf(v);
            }
        }
}

// ---------------------------------------------------------------------------
// Input-projection GEMM for LSTM gates. M = 256 (2 dirs x 128 gates).
// Gin output layout: [d][t][b][128] fp32 (gate-major for fast recurrent reads).
// nmajor=0: Xin is [Cin][N]; nmajor=1: Xin is [N][Cin].
// ---------------------------------------------------------------------------
__global__ __launch_bounds__(256) void proj_gemm_kernel(
        const unsigned short* __restrict__ Wp,   // [256][Cin] bf16
        const unsigned short* __restrict__ Xin,
        float* __restrict__ Gin,                 // [2][1024][32][128]
        const float* __restrict__ bias,          // [256]
        int Cin, int nmajor) {
    const int N = NPOS;
    const int M = 256;
    __shared__ __align__(16) unsigned short As[128 * 32];
    __shared__ __align__(16) unsigned short Bs[128 * 32];   // [n][k]

    int tid  = threadIdx.x;
    int lane = tid & 31;
    int wave = tid >> 5;
    int waveM = wave >> 1;
    int waveN = wave & 1;
    int m0 = blockIdx.y * 128;
    int n0 = blockIdx.x * 128;

    v8f acc[2][4];
    #pragma unroll
    for (int mt = 0; mt < 2; ++mt)
        #pragma unroll
        for (int nt = 0; nt < 4; ++nt) acc[mt][nt] = (v8f){};

    const int ar = tid >> 1;
    const int ak = (tid & 1) * 16;

    for (int cb = 0; cb < Cin; cb += 32) {
        const f32x4* ws = (const f32x4*)(Wp + ((size_t)(m0 + ar) * Cin + cb + ak));
        f32x4 wa = ws[0], wb = ws[1];

        f32x4 xv0, xv1;
        unsigned short tmp[16];
        if (nmajor) {   // contiguous copy: Bs[n][k] <- Xin[n0+ar][cb+ak..]
            const f32x4* xs = (const f32x4*)(Xin + ((size_t)(n0 + ar) * Cin + cb + ak));
            xv0 = xs[0]; xv1 = xs[1];
        } else {        // transpose: 16 cols at row cb+(tid&31)
            int bk = tid & 31, bseg = (tid >> 5) * 16;
            const unsigned short* xrow = Xin + (size_t)(cb + bk) * N + n0 + bseg;
            union { f32x4 f[2]; unsigned short u[16]; } bu;
            bu.f[0] = ((const f32x4*)xrow)[0];
            bu.f[1] = ((const f32x4*)xrow)[1];
            #pragma unroll
            for (int i = 0; i < 16; ++i) tmp[i] = bu.u[i];
        }

        __syncthreads();
        ((f32x4*)(As + ar * 32 + ak))[0] = wa;
        ((f32x4*)(As + ar * 32 + ak))[1] = wb;
        if (nmajor) {
            ((f32x4*)(Bs + ar * 32 + ak))[0] = xv0;
            ((f32x4*)(Bs + ar * 32 + ak))[1] = xv1;
        } else {
            int bk = tid & 31, bseg = (tid >> 5) * 16;
            #pragma unroll
            for (int i = 0; i < 16; ++i) Bs[(bseg + i) * 32 + bk] = tmp[i];
        }
        __syncthreads();

        const int lo = (lane < 16) ? 0 : 16;
        const int ko = (lane < 16) ? 0 : 16;
        Frag afr[2];
        #pragma unroll
        for (int mt = 0; mt < 2; ++mt) {
            int m = waveM * 32 + mt * 16 + (lane & 15);
            afr[mt].f4[0] = *(const f32x4*)(As + m * 32 + lo);
            afr[mt].f4[1] = *(const f32x4*)(As + m * 32 + lo + 8);
        }
        #pragma unroll
        for (int nt = 0; nt < 4; ++nt) {
            Frag bfr;
            int n = waveN * 64 + nt * 16 + (lane & 15);
            bfr.f4[0] = *(const f32x4*)(Bs + n * 32 + ko);
            bfr.f4[1] = *(const f32x4*)(Bs + n * 32 + ko + 8);
            #pragma unroll
            for (int mt = 0; mt < 2; ++mt)
                acc[mt][nt] = wmma_bf16(afr[mt], bfr, acc[mt][nt]);
        }
    }

    // epilogue: gates transposed to [d][t][b][128]
    #pragma unroll
    for (int mt = 0; mt < 2; ++mt)
        #pragma unroll
        for (int nt = 0; nt < 4; ++nt) {
            int gn = n0 + waveN * 64 + nt * 16 + (lane & 15);
            int b = gn >> 10, t = gn & 1023;
            #pragma unroll
            for (int r = 0; r < 8; ++r) {
                int gm = m0 + waveM * 32 + mt * 16 + ((lane < 16) ? r : 8 + r);
                int d = gm >> 7, g = gm & 127;
                Gin[(((size_t)d * 1024 + t) * 32 + b) * 128 + g] = acc[mt][nt][r] + bias[gm];
            }
        }
}

// ---------------------------------------------------------------------------
// Recurrent bidirectional LSTM. grid=2 (direction), block=256.
// gates(32b x 128g) = WMMA(h @ Whh^T) + Gin[d][t][b][g]; cell state in regs.
// ---------------------------------------------------------------------------
__global__ __launch_bounds__(256) void lstm_kernel(
        const float* __restrict__ Gin,      // [2][1024][32][128]
        const float* __restrict__ Whh,      // [2][128][32] fp32
        unsigned short* __restrict__ Sout,  // layer0: [NPOS][64] bf16 (n-major)
        unsigned short* __restrict__ Pout,  // layer1: [2][NPOS][32] bf16
        float* __restrict__ Fout,           // layer1: [2][NPOS][32] fp32
        int isLayer1) {
    const int N = NPOS;
    int d    = blockIdx.x;
    int tid  = threadIdx.x;
    int lane = tid & 31;
    int wave = tid >> 5;   // ntile

    __shared__ __align__(16) unsigned short Hs[32 * 32];  // [b][j] bf16
    __shared__ __align__(16) float Gs[32 * 128];          // recurrent gate partials

    for (int i = tid; i < 32 * 32; i += 256) Hs[i] = 0;

    Frag bfr;   // Whh tile, resident all 1024 steps
    {
        int g  = wave * 16 + (lane & 15);
        int ko = (lane < 16) ? 0 : 16;
        const float* wr = Whh + ((size_t)d * 128 + g) * 32 + ko;
        #pragma unroll
        for (int e = 0; e < 16; ++e) bfr.u[e] = f2bf(wr[e]);
    }

    const int ub = (tid * 4) >> 5;   // batch row for this thread's 4 pairs
    const int uj = (tid * 4) & 31;   // first hidden index (4 consecutive)
    float cst[4] = {0.f, 0.f, 0.f, 0.f};
    __syncthreads();

    const int lo = (lane < 16) ? 0 : 16;
    const int mloc = (lane & 15);
    const int gcol = wave * 16 + (lane & 15);

    for (int tt = 0; tt < 1024; ++tt) {
        int t = d ? (1023 - tt) : tt;

        Frag a0, a1;
        a0.f4[0] = *(const f32x4*)(Hs + mloc * 32 + lo);
        a0.f4[1] = *(const f32x4*)(Hs + mloc * 32 + lo + 8);
        a1.f4[0] = *(const f32x4*)(Hs + (16 + mloc) * 32 + lo);
        a1.f4[1] = *(const f32x4*)(Hs + (16 + mloc) * 32 + lo + 8);

        v8f z = (v8f){};
        v8f r0 = wmma_bf16(a0, bfr, z);
        v8f r1 = wmma_bf16(a1, bfr, z);

        #pragma unroll
        for (int r = 0; r < 8; ++r) {
            int m = (lane < 16) ? r : 8 + r;
            Gs[m * 128 + gcol]        = r0[r];
            Gs[(16 + m) * 128 + gcol] = r1[r];
        }
        __syncthreads();

        {
            const float* gbase = Gin + (((size_t)d * 1024 + t) * 32 + ub) * 128 + uj;
            f32x4 gI = *(const f32x4*)(gbase);
            f32x4 gF = *(const f32x4*)(gbase + 32);
            f32x4 gG = *(const f32x4*)(gbase + 64);
            f32x4 gO = *(const f32x4*)(gbase + 96);
            const float* sbase = Gs + ub * 128 + uj;
            f32x4 sI = *(const f32x4*)(sbase);
            f32x4 sF = *(const f32x4*)(sbase + 32);
            f32x4 sG = *(const f32x4*)(sbase + 64);
            f32x4 sO = *(const f32x4*)(sbase + 96);
            size_t n = (size_t)ub * 1024 + t;
            #pragma unroll
            for (int p = 0; p < 4; ++p) {
                float c = sigf(sF[p] + gF[p]) * cst[p] + sigf(sI[p] + gI[p]) * tanhf(sG[p] + gG[p]);
                float h = sigf(sO[p] + gO[p]) * tanhf(c);
                cst[p] = c;
                Hs[ub * 32 + uj + p] = f2bf(h);
                if (!isLayer1) {
                    Sout[n * 64 + d * 32 + uj + p] = f2bf(h);
                } else {
                    size_t o = ((size_t)d * N + n) * 32 + uj + p;
                    Pout[o] = f2bf(h);
                    Fout[o] = h;
                }
            }
        }
        __syncthreads();
    }
}

// ---------------------------------------------------------------------------
// VQ nearest-code search: score = x.c - 0.5||c||^2, argmax via WMMA + shfl.
// ---------------------------------------------------------------------------
__global__ __launch_bounds__(256) void vq_argmax_kernel(
        const unsigned short* __restrict__ P, const unsigned short* __restrict__ CBb,
        const float* __restrict__ cnorm, int* __restrict__ idxOut) {
    const int N = NPOS;
    int d    = blockIdx.y;
    int tid  = threadIdx.x;
    int lane = tid & 31;
    int wave = tid >> 5;
    int p0 = (blockIdx.x * 8 + wave) * 16;

    Frag a;
    {
        int p  = p0 + (lane & 15);
        int lo = (lane < 16) ? 0 : 16;
        const unsigned short* row = P + ((size_t)d * N + p) * 32;
        a.f4[0] = *(const f32x4*)(row + lo);
        a.f4[1] = *(const f32x4*)(row + lo + 8);
    }

    float best[8];
    int   bidx[8];
    #pragma unroll
    for (int r = 0; r < 8; ++r) { best[r] = -3.0e38f; bidx[r] = 0; }

    int ko = (lane < 16) ? 0 : 16;
    for (int nt = 0; nt < 32; ++nt) {
        Frag b;
        int c = nt * 16 + (lane & 15);
        const unsigned short* cr = CBb + ((size_t)d * 512 + c) * 32 + ko;
        b.f4[0] = *(const f32x4*)cr;
        b.f4[1] = *(const f32x4*)(cr + 8);
        v8f accz = (v8f){};
        v8f acc = wmma_bf16(a, b, accz);
        float cn = cnorm[d * 512 + c];
        #pragma unroll
        for (int r = 0; r < 8; ++r) {
            float s = acc[r] - cn;
            if (s > best[r]) { best[r] = s; bidx[r] = c; }
        }
    }

    #pragma unroll
    for (int r = 0; r < 8; ++r) {
        #pragma unroll
        for (int off = 8; off; off >>= 1) {
            float ob = __shfl_xor(best[r], off, 16);
            int   oi = __shfl_xor(bidx[r], off, 16);
            if (ob > best[r] || (ob == best[r] && oi < bidx[r])) { best[r] = ob; bidx[r] = oi; }
        }
    }
    if ((lane & 15) == 0) {
        #pragma unroll
        for (int r = 0; r < 8; ++r) {
            int m = (lane < 16) ? r : 8 + r;
            idxOut[(size_t)d * N + p0 + m] = bidx[r];
        }
    }
}

// ---------------------------------------------------------------------------
// Emit subsampled codes + per-block loss partials (deterministic).
// ---------------------------------------------------------------------------
__global__ __launch_bounds__(256) void vq_finalize_kernel(
        const float* __restrict__ F, const int* __restrict__ idx,
        const float* __restrict__ cb_f, const float* __restrict__ cb_b,
        float* __restrict__ out, float* __restrict__ partial) {
    const int N = NPOS;
    int gid = blockIdx.x * 256 + threadIdx.x;   // 0..65535
    int d = gid >> 15, p = gid & 32767;
    int b = p >> 10, t = p & 1023;
    const float* cb = d ? cb_b : cb_f;
    int q = idx[(size_t)d * N + p];
    const float* qv = cb + (size_t)q * 32;
    const float* xv = F + ((size_t)d * N + p) * 32;
    bool wsel = (d == 0) ? ((t & 31) == 0) : ((t & 31) == 31);
    size_t ob = ((size_t)b * 32 + (t >> 5)) * 64 + (d ? 32 : 0);
    float s = 0.0f;
    #pragma unroll
    for (int j = 0; j < 32; ++j) {
        float dq = xv[j] - qv[j];
        s += dq * dq;
        if (wsel) out[ob + j] = qv[j];
    }
    __shared__ float red[256];
    red[threadIdx.x] = s;
    __syncthreads();
    for (int st = 128; st; st >>= 1) {
        if (threadIdx.x < st) red[threadIdx.x] += red[threadIdx.x + st];
        __syncthreads();
    }
    if (threadIdx.x == 0) partial[blockIdx.x] = red[0];
}

__global__ __launch_bounds__(256) void loss_reduce_kernel(const float* __restrict__ partial,
                                                          float* __restrict__ out) {
    __shared__ float red[256];
    red[threadIdx.x] = partial[threadIdx.x];
    __syncthreads();
    for (int st = 128; st; st >>= 1) {
        if (threadIdx.x < st) red[threadIdx.x] += red[threadIdx.x + st];
        __syncthreads();
    }
    if (threadIdx.x == 0) out[65536] = 0.25f * red[0] / 1048576.0f;
}

// ---------------------------------------------------------------------------
// Host launch
// ---------------------------------------------------------------------------
extern "C" void kernel_launch(void* const* d_in, const int* in_sizes, int n_in,
                              void* d_out, int out_size, void* d_ws, size_t ws_size,
                              hipStream_t stream) {
    (void)in_sizes; (void)n_in; (void)out_size; (void)ws_size;
    const int N = NPOS;

    const float* x        = (const float*)d_in[0];
    const float* c_src    = (const float*)d_in[1];
    const float* conv_w[3] = {(const float*)d_in[2],  (const float*)d_in[8],  (const float*)d_in[14]};
    const float* conv_b[3] = {(const float*)d_in[3],  (const float*)d_in[9],  (const float*)d_in[15]};
    const float* bn_g[3]   = {(const float*)d_in[4],  (const float*)d_in[10], (const float*)d_in[16]};
    const float* bn_b[3]   = {(const float*)d_in[5],  (const float*)d_in[11], (const float*)d_in[17]};
    const float* bn_m[3]   = {(const float*)d_in[6],  (const float*)d_in[12], (const float*)d_in[18]};
    const float* bn_v[3]   = {(const float*)d_in[7],  (const float*)d_in[13], (const float*)d_in[19]};
    const float* lstm0_wih = (const float*)d_in[20];
    const float* lstm0_whh = (const float*)d_in[21];
    const float* lstm0_b   = (const float*)d_in[22];
    const float* lstm1_wih = (const float*)d_in[23];
    const float* lstm1_whh = (const float*)d_in[24];
    const float* lstm1_b   = (const float*)d_in[25];
    const float* cb_f      = (const float*)d_in[26];
    const float* cb_b      = (const float*)d_in[27];

    char* base = (char*)d_ws;
    size_t off = 0;
    auto alloc = [&](size_t bytes) -> void* {
        void* p = base + off;
        off += (bytes + 255) & ~(size_t)255;
        return p;
    };

    unsigned short* bufA  = (unsigned short*)alloc((size_t)512 * N * 2);
    unsigned short* bufB  = (unsigned short*)alloc((size_t)512 * N * 2);
    unsigned short* Wp0   = (unsigned short*)alloc((size_t)5 * 512 * 352 * 2);
    unsigned short* Wp1   = (unsigned short*)alloc((size_t)5 * 512 * 512 * 2);
    unsigned short* Wp2   = (unsigned short*)alloc((size_t)5 * 512 * 512 * 2);
    unsigned short* Wih0p = (unsigned short*)alloc((size_t)256 * 512 * 2);
    unsigned short* Wih1p = (unsigned short*)alloc((size_t)256 * 64 * 2);
    float* scale0 = (float*)alloc(512 * 4);  float* bias0 = (float*)alloc(512 * 4);
    float* scale1 = (float*)alloc(512 * 4);  float* bias1 = (float*)alloc(512 * 4);
    float* scale2 = (float*)alloc(512 * 4);  float* bias2 = (float*)alloc(512 * 4);
    float* biasL0 = (float*)alloc(256 * 4);
    float* biasL1 = (float*)alloc(256 * 4);
    float* Gin    = (float*)alloc((size_t)256 * N * 4);   // [2][1024][32][128]
    unsigned short* S0  = (unsigned short*)alloc((size_t)N * 64 * 2);  // [n][64]
    unsigned short* P   = (unsigned short*)alloc((size_t)2 * N * 32 * 2);
    float* F            = (float*)alloc((size_t)2 * N * 32 * 4);
    unsigned short* CBb = (unsigned short*)alloc((size_t)2 * 512 * 32 * 2);
    float* cnorm        = (float*)alloc(1024 * 4);
    int* vqidx          = (int*)alloc((size_t)2 * N * 4);
    float* lossPartial  = (float*)alloc(256 * 4);
    float* out = (float*)d_out;

    // ---- pack / fold ----
    pack_x0_kernel<<<(int)(((size_t)352 * N + 255) / 256), 256, 0, stream>>>(x, c_src, bufA);
    pack_convw_kernel<<<(int)(((size_t)5 * 512 * 352 + 255) / 256), 256, 0, stream>>>(conv_w[0], Wp0, 512, 336, 352);
    pack_convw_kernel<<<(int)(((size_t)5 * 512 * 512 + 255) / 256), 256, 0, stream>>>(conv_w[1], Wp1, 512, 512, 512);
    pack_convw_kernel<<<(int)(((size_t)5 * 512 * 512 + 255) / 256), 256, 0, stream>>>(conv_w[2], Wp2, 512, 512, 512);
    bnfold_kernel<<<2, 256, 0, stream>>>(conv_b[0], bn_g[0], bn_b[0], bn_m[0], bn_v[0], scale0, bias0);
    bnfold_kernel<<<2, 256, 0, stream>>>(conv_b[1], bn_g[1], bn_b[1], bn_m[1], bn_v[1], scale1, bias1);
    bnfold_kernel<<<2, 256, 0, stream>>>(conv_b[2], bn_g[2], bn_b[2], bn_m[2], bn_v[2], scale2, bias2);
    pack_wih_kernel<<<(256 * 512 + 255) / 256, 256, 0, stream>>>(lstm0_wih, lstm0_b, Wih0p, biasL0, 512);
    pack_wih_kernel<<<(256 * 64 + 255) / 256, 256, 0, stream>>>(lstm1_wih, lstm1_b, Wih1p, biasL1, 64);
    pack_cb_kernel<<<(2 * 512 * 32 + 255) / 256, 256, 0, stream>>>(cb_f, cb_b, CBb, cnorm);

    // ---- conv stack (haloed implicit-GEMM WMMA, BN+ReLU fused) ----
    dim3 g512(N / 128, 4), g256(N / 128, 2);
    conv_gemm_kernel<<<g512, 256, 0, stream>>>(Wp0, bufA, bufB, scale0, bias0, 512, 352);
    conv_gemm_kernel<<<g512, 256, 0, stream>>>(Wp1, bufB, bufA, scale1, bias1, 512, 512);
    conv_gemm_kernel<<<g512, 256, 0, stream>>>(Wp2, bufA, bufB, scale2, bias2, 512, 512);

    // ---- LSTM 0 ----
    proj_gemm_kernel<<<g256, 256, 0, stream>>>(Wih0p, bufB, Gin, biasL0, 512, 0);
    lstm_kernel<<<2, 256, 0, stream>>>(Gin, lstm0_whh, S0, nullptr, nullptr, 0);

    // ---- LSTM 1 ----
    proj_gemm_kernel<<<g256, 256, 0, stream>>>(Wih1p, S0, Gin, biasL1, 64, 1);
    lstm_kernel<<<2, 256, 0, stream>>>(Gin, lstm1_whh, nullptr, P, F, 1);

    // ---- VQ ----
    vq_argmax_kernel<<<dim3(256, 2), 256, 0, stream>>>(P, CBb, cnorm, vqidx);
    vq_finalize_kernel<<<256, 256, 0, stream>>>(F, vqidx, cb_f, cb_b, out, lossPartial);
    loss_reduce_kernel<<<1, 256, 0, stream>>>(lossPartial, out);
}
